// RWKV7Attention_55697135895325
// MI455X (gfx1250) — matrix operand order repaired and checked by reading, not compiled, not run on hardware
//
#include <hip/hip_runtime.h>
#include <hip/hip_bf16.h>
#include <math.h>

// Problem constants (match reference)
#define BB 2
#define TTK 4096
#define CCK 2048
#define HHK 32
#define DDK 64
#define NNK (BB*TTK)              // 8192 tokens
#define NCK ((size_t)NNK*CCK)     // 16,777,216 elements
#define RWK 64
#define RAK 64
#define RGK 128
#define RVK 32
#define GN_EPS_K (64.0f*1e-5f)

typedef __attribute__((ext_vector_type(16))) __bf16 v16bf;
typedef __attribute__((ext_vector_type(16))) unsigned short v16us;
typedef __attribute__((ext_vector_type(8)))  float v8f;

__device__ __forceinline__ unsigned short f2bf(float f) {
  unsigned int x = __float_as_uint(f);
  x += 0x7FFFu + ((x >> 16) & 1u);   // round-to-nearest-even
  return (unsigned short)(x >> 16);
}
__device__ __forceinline__ float fast_rcp(float x) { return __builtin_amdgcn_rcpf(x); }
__device__ __forceinline__ float sigmoidf_(float x) {
  return fast_rcp(1.f + __expf(-x));
}
__device__ __forceinline__ float tanhf_(float x) {
  float xc = fminf(fmaxf(x, -15.f), 15.f);
  float e = __expf(2.f * xc);
  return (e - 1.f) * fast_rcp(e + 1.f);
}

// ---------------- f32 -> bf16 weight conversion ----------------
__global__ void f32_to_bf16_kernel(const float* __restrict__ src,
                                   unsigned short* __restrict__ dst, size_t n) {
  size_t i = (size_t)blockIdx.x * blockDim.x + threadIdx.x;
  size_t stride = (size_t)gridDim.x * blockDim.x;
  for (; i < n; i += stride) dst[i] = f2bf(src[i]);
}

// ---------------- Generic WMMA bf16 GEMM:  D = epi( A @ W^T ) ----------------
// W is bf16 [Ncols][K] (row = output channel), so D[m][n] = sum_k A[m][k]*W[n][k].
// Each wave computes a 16 x (NT*16) output block: ONE A fragment per K-step feeds
// NT back-to-back WMMAs. K loop is software-pipelined (double-buffered A and B)
// so global-load latency hides behind the WMMA issue; global_prefetch reaches
// 4 K-steps ahead.
// MIX mode: A[m][k] = x[m][k] + (x[m-1][k] - x[m][k]) * coef[k], with x[-1]=0 at t%T==0.
#define EPI_NONE  0
#define EPI_TANH  1
#define EPI_SIG   2
#define EPI_SIGB  3   // sigmoid(val + bias[col])
#define EPI_WDEC  4   // decay = exp(-exp(-softplus(-(val+bias)) - 0.5))
#define EPI_VLERP 5   // gv = sigmoid(val+bias); D = v + gv*(v_first - v)   (aux0=v, aux1=v_first)

template <bool MIX>
__device__ __forceinline__ v16us buildA(const float* __restrict__ xr,
                                        const float* __restrict__ xp,
                                        const float* __restrict__ coef,
                                        bool hasPrev, int kb, int kh) {
  // k order used here for A (and B): element i holds K = kh*16 + i
  v16us au;
  #pragma unroll
  for (int h2 = 0; h2 < 2; ++h2) {
    const int kbase = kb + kh * 16 + h2 * 8;
    #pragma unroll
    for (int e = 0; e < 8; ++e) {
      float xc = xr[kbase + e];
      float val;
      if (MIX) {
        float xs = xp[kbase + e];
        if (!hasPrev) xs = 0.f;
        val = xc + (xs - xc) * coef[kbase + e];
      } else {
        val = xc;
      }
      au[h2 * 8 + e] = f2bf(val);
    }
  }
  return au;
}

template <bool MIX, int NT>
__global__ void gemm_bf16_wmma(const float* __restrict__ A,
                               const float* __restrict__ coef,
                               const unsigned short* __restrict__ W,
                               float* __restrict__ Dst,
                               const float* __restrict__ bias,
                               const float* __restrict__ aux0,
                               const float* __restrict__ aux1,
                               int M, int K, int Ncols, int EPI) {
  const int lane = threadIdx.x & 31;
  const int wave = threadIdx.x >> 5;
  const int m    = lane & 15;   // row-in-tile (A) / col-in-tile (B,D)
  const int kh   = lane >> 4;   // K-half selector (contiguous k order used here)
  const int ngroups = (Ncols >> 4) / NT;     // N-tile groups of NT
  const int grp = blockIdx.x * 4 + wave;
  if (grp >= (M >> 4) * ngroups) return;     // wave-uniform exit (EXEC stays all-ones)
  const int mtile  = grp / ngroups;
  const int ngrp   = grp % ngroups;
  const int ncol0  = ngrp * NT * 16;

  const int rowA = mtile * 16 + m;
  const bool hasPrev = MIX ? ((rowA % TTK) != 0) : false;
  const float* xr = A + (size_t)rowA * K;
  const float* xp = hasPrev ? (xr - K) : xr;       // always a valid address
  const unsigned short* wr[NT];
  #pragma unroll
  for (int s = 0; s < NT; ++s)
    wr[s] = W + (size_t)(ncol0 + s * 16 + m) * K;

  v8f acc[NT];
  #pragma unroll
  for (int s = 0; s < NT; ++s) acc[s] = (v8f){};

  // pipeline prologue: fragments for kb = 0
  v16us auCur = buildA<MIX>(xr, xp, coef, hasPrev, 0, kh);
  v16us buCur[NT];
  #pragma unroll
  for (int s = 0; s < NT; ++s)
    buCur[s] = *(const v16us*)(wr[s] + kh * 16);

  for (int kb = 0; kb < K; kb += 32) {
    const int kn = (kb + 32 < K) ? (kb + 32) : kb;   // last iter: redundant safe reload
    // deep prefetch (global_prefetch_b8), 4 K-steps ahead
    if (kb + 128 < K) {
      __builtin_prefetch((const void*)(xr + kb + 128 + kh * 8), 0, 3);
      #pragma unroll
      for (int s = 0; s < NT; ++s)
        __builtin_prefetch((const void*)(wr[s] + kb + 128 + kh * 16), 0, 3);
    }
    // issue next-step loads before consuming current fragments
    v16us auNext = buildA<MIX>(xr, xp, coef, hasPrev, kn, kh);
    v16us buNext[NT];
    #pragma unroll
    for (int s = 0; s < NT; ++s)
      buNext[s] = *(const v16us*)(wr[s] + kn + kh * 16);

    v16bf av = __builtin_bit_cast(v16bf, auCur);
    #pragma unroll
    for (int s = 0; s < NT; ++s) {
      v16bf bv = __builtin_bit_cast(v16bf, buCur[s]);
      acc[s] = __builtin_amdgcn_wmma_f32_16x16x32_bf16(false, av, false, bv,
                                                       (short)0, acc[s], false, false);
    }
    auCur = auNext;
    #pragma unroll
    for (int s = 0; s < NT; ++s) buCur[s] = buNext[s];
  }

  // C/D layout: lane n = lane&15, row = (lane>>4)*8 + j
  #pragma unroll
  for (int s = 0; s < NT; ++s) {
    const int ocol = ncol0 + s * 16 + m;
    float bcol = 0.f;
    if (EPI == EPI_SIGB || EPI == EPI_WDEC || EPI == EPI_VLERP) bcol = bias[ocol];
    #pragma unroll
    for (int j = 0; j < 8; ++j) {
      const int orow = mtile * 16 + kh * 8 + j;
      const size_t oidx = (size_t)orow * Ncols + ocol;
      float val = acc[s][j];
      if (EPI == EPI_TANH) {
        val = tanhf_(val);
      } else if (EPI == EPI_SIG) {
        val = sigmoidf_(val);
      } else if (EPI == EPI_SIGB) {
        val = sigmoidf_(val + bcol);
      } else if (EPI == EPI_WDEC) {
        float wpre = val + bcol;
        float z = -wpre;
        float sp = (z > 20.f) ? z : log1pf(__expf(z));  // softplus(-wpre)
        val = __expf(-__expf(-sp - 0.5f));              // decay in (0,1)
      } else if (EPI == EPI_VLERP) {
        float gv = sigmoidf_(val + bcol);
        float vv = aux0[oidx];
        val = vv + gv * (aux1[oidx] - vv);
      }
      Dst[oidx] = val;
    }
  }
}

// ---------------- kk normalization + k modification ----------------
// One 64-thread block per (token, head).
__global__ void prep_scan(float* __restrict__ k,          // in: k, out: k_mod
                          const float* __restrict__ ag,   // a_gate
                          const float* __restrict__ k_k,
                          const float* __restrict__ k_a,
                          float* __restrict__ kkout) {
  __shared__ float red[DDK];
  const int tid = threadIdx.x;
  const int n = blockIdx.x / HHK;
  const int h = blockIdx.x % HHK;
  const int c = h * DDK + tid;
  const size_t idx = (size_t)n * CCK + c;
  float kv = k[idx];
  float kkr = kv * k_k[c];
  red[tid] = kkr * kkr; __syncthreads();
  #pragma unroll
  for (int s = 32; s > 0; s >>= 1) { if (tid < s) red[tid] += red[tid + s]; __syncthreads(); }
  float nrm = fmaxf(sqrtf(red[0]), 1e-12f);
  kkout[idx] = kkr / nrm;
  float a = ag[idx];
  k[idx] = kv + kv * (a - 1.f) * k_a[c];
}

// ---------------- WKV7 linear-RNN scan ----------------
// One 64-thread block per (b,h); thread i owns value-row i of the 64x64 state
// entirely in VGPRs (no cross-lane reductions needed). Per-step vectors are
// staged to LDS once (packed float4) and broadcast-read. Global loads for the
// next timestep are issued before the compute of the current one.
__global__ void wkv_scan(const float* __restrict__ r, const float* __restrict__ k,
                         const float* __restrict__ v, const float* __restrict__ dec,
                         const float* __restrict__ kk, const float* __restrict__ ag,
                         const float* __restrict__ s0,
                         float* __restrict__ y4, float* __restrict__ sT) {
  const int i = threadIdx.x;
  const int b = blockIdx.x / HHK;
  const int h = blockIdx.x % HHK;
  __shared__ float4 Lp[DDK];   // {decay, b=kk*ag, k, r}
  __shared__ float  La[DDK];   // a = -kk
  __shared__ float  Lv[DDK];

  float S[DDK];
  {
    const float* s0p = s0 + ((size_t)(b * HHK + h) * DDK + i) * DDK;
    #pragma unroll
    for (int j = 0; j < DDK; ++j) S[j] = s0p[j];
  }

  float kkv, agv, kv, rv, dv, vv;
  auto loadstep = [&](int t) {
    const size_t base = ((size_t)(b * TTK + t)) * CCK + h * DDK + i;
    kkv = kk[base]; agv = ag[base]; kv = k[base];
    rv = r[base];   dv = dec[base]; vv = v[base];
  };
  loadstep(0);

  for (int t = 0; t < TTK; ++t) {
    const size_t base = ((size_t)(b * TTK + t)) * CCK + h * DDK;
    Lp[i] = make_float4(dv, kkv * agv, kv, rv);
    La[i] = -kkv;
    Lv[i] = vv;
    __syncthreads();
    if (t + 1 < TTK) loadstep(t + 1);   // overlap with compute below

    float Sa = 0.f;
    #pragma unroll
    for (int j = 0; j < DDK; ++j) Sa += S[j] * La[j];
    const float vi = Lv[i];
    float y = 0.f;
    #pragma unroll
    for (int j = 0; j < DDK; ++j) {
      float4 p = Lp[j];
      float s = S[j] * p.x + Sa * p.y + vi * p.z;
      S[j] = s;
      y += s * p.w;
    }
    y4[base + i] = y;
    __syncthreads();
  }

  float* sTp = sT + ((size_t)(b * HHK + h) * DDK + i) * DDK;
  #pragma unroll
  for (int j = 0; j < DDK; ++j) sTp[j] = S[j];
}

// ---------------- GroupNorm(per head) + correction + gate ----------------
__global__ void gn_corr_gate(const float* __restrict__ y4, const float* __restrict__ r,
                             const float* __restrict__ k, const float* __restrict__ v,
                             const float* __restrict__ g, const float* __restrict__ r_k,
                             const float* __restrict__ gnw, const float* __restrict__ gnb,
                             float* __restrict__ out) {
  __shared__ float red[DDK];
  const int tid = threadIdx.x;
  const int n = blockIdx.x / HHK;
  const int h = blockIdx.x % HHK;
  const int c = h * DDK + tid;
  const size_t idx = (size_t)n * CCK + c;

  float yv = y4[idx];
  red[tid] = yv; __syncthreads();
  #pragma unroll
  for (int s = 32; s > 0; s >>= 1) { if (tid < s) red[tid] += red[tid + s]; __syncthreads(); }
  float mu = red[0] * (1.f / DDK); __syncthreads();

  float dv = yv - mu;
  red[tid] = dv * dv; __syncthreads();
  #pragma unroll
  for (int s = 32; s > 0; s >>= 1) { if (tid < s) red[tid] += red[tid + s]; __syncthreads(); }
  float var = red[0] * (1.f / DDK); __syncthreads();
  float yn = dv * rsqrtf(var + GN_EPS_K) * gnw[c] + gnb[c];

  red[tid] = r[idx] * k[idx] * r_k[c]; __syncthreads();
  #pragma unroll
  for (int s = 32; s > 0; s >>= 1) { if (tid < s) red[tid] += red[tid + s]; __syncthreads(); }
  float cs = red[0];

  out[idx] = (yn + cs * v[idx]) * g[idx];
}

// ---------------- host launcher ----------------
extern "C" void kernel_launch(void* const* d_in, const int* in_sizes, int n_in,
                              void* d_out, int out_size, void* d_ws, size_t ws_size,
                              hipStream_t stream) {
  (void)in_sizes; (void)n_in; (void)out_size; (void)ws_size;
  // inputs: x, v_first, s0, then params in dict order
  const float* x      = (const float*)d_in[0];
  const float* vfirst = (const float*)d_in[1];
  const float* s0     = (const float*)d_in[2];
  const float* x_r = (const float*)d_in[3];
  const float* x_w = (const float*)d_in[4];
  const float* x_k = (const float*)d_in[5];
  const float* x_v = (const float*)d_in[6];
  const float* x_a = (const float*)d_in[7];
  const float* x_g = (const float*)d_in[8];
  const float* k_k = (const float*)d_in[9];
  const float* k_a = (const float*)d_in[10];
  const float* r_k = (const float*)d_in[11];
  const float* Wr  = (const float*)d_in[12];
  const float* Wk  = (const float*)d_in[13];
  const float* Wv  = (const float*)d_in[14];
  const float* Wo  = (const float*)d_in[15];
  const float* w0  = (const float*)d_in[16];
  const float* w1  = (const float*)d_in[17];
  const float* w2  = (const float*)d_in[18];
  const float* a0  = (const float*)d_in[19];
  const float* a1  = (const float*)d_in[20];
  const float* a2  = (const float*)d_in[21];
  const float* g1  = (const float*)d_in[22];
  const float* g2  = (const float*)d_in[23];
  const float* v0  = (const float*)d_in[24];
  const float* v1  = (const float*)d_in[25];
  const float* v2  = (const float*)d_in[26];
  const float* gnw = (const float*)d_in[27];
  const float* gnb = (const float*)d_in[28];

  float* outY  = (float*)d_out;            // (B,T,C)
  float* outVF = outY + NCK;               // (B,T,C) pass-through
  float* outST = outY + 2 * NCK;           // (B,H,D,D)

  // bump allocator on workspace
  char* cur = (char*)d_ws;
  auto alloc = [&](size_t bytes) -> void* {
    void* p = (void*)cur;
    cur += (bytes + 255) & ~(size_t)255;
    return p;
  };

  // bf16 weight copies
  unsigned short* WrB = (unsigned short*)alloc((size_t)CCK * CCK * 2);
  unsigned short* WkB = (unsigned short*)alloc((size_t)CCK * CCK * 2);
  unsigned short* WvB = (unsigned short*)alloc((size_t)CCK * CCK * 2);
  unsigned short* WoB = (unsigned short*)alloc((size_t)CCK * CCK * 2);
  unsigned short* w2B = (unsigned short*)alloc((size_t)RWK * CCK * 2);
  unsigned short* w1B = (unsigned short*)alloc((size_t)CCK * RWK * 2);
  unsigned short* a2B = (unsigned short*)alloc((size_t)RAK * CCK * 2);
  unsigned short* a1B = (unsigned short*)alloc((size_t)CCK * RAK * 2);
  unsigned short* g2B = (unsigned short*)alloc((size_t)RGK * CCK * 2);
  unsigned short* g1B = (unsigned short*)alloc((size_t)CCK * RGK * 2);
  unsigned short* v2B = (unsigned short*)alloc((size_t)RVK * CCK * 2);
  unsigned short* v1B = (unsigned short*)alloc((size_t)CCK * RVK * 2);

  // f32 activation buffers
  float* rbuf   = (float*)alloc(NCK * 4);
  float* kbuf   = (float*)alloc(NCK * 4);
  float* vbuf   = (float*)alloc(NCK * 4);
  float* decbuf = (float*)alloc(NCK * 4);   // decay; reused as gated-y after scan
  float* kkbuf  = (float*)alloc(NCK * 4);
  float* agbuf  = (float*)alloc(NCK * 4);
  float* gbuf   = (float*)alloc(NCK * 4);
  float* y4buf  = (float*)alloc(NCK * 4);
  float* wh  = (float*)alloc((size_t)NNK * RWK * 4);
  float* ah  = (float*)alloc((size_t)NNK * RAK * 4);
  float* gh  = (float*)alloc((size_t)NNK * RGK * 4);
  float* vgh = (float*)alloc((size_t)NNK * RVK * 4);

  auto cvt = [&](const float* s, unsigned short* dd, size_t n) {
    int blocks = (int)((n + 256 * 8 - 1) / (256 * 8));
    if (blocks < 1) blocks = 1;
    f32_to_bf16_kernel<<<blocks, 256, 0, stream>>>(s, dd, n);
  };
  cvt(Wr, WrB, (size_t)CCK * CCK); cvt(Wk, WkB, (size_t)CCK * CCK);
  cvt(Wv, WvB, (size_t)CCK * CCK); cvt(Wo, WoB, (size_t)CCK * CCK);
  cvt(w2, w2B, (size_t)RWK * CCK); cvt(w1, w1B, (size_t)CCK * RWK);
  cvt(a2, a2B, (size_t)RAK * CCK); cvt(a1, a1B, (size_t)CCK * RAK);
  cvt(g2, g2B, (size_t)RGK * CCK); cvt(g1, g1B, (size_t)CCK * RGK);
  cvt(v2, v2B, (size_t)RVK * CCK); cvt(v1, v1B, (size_t)CCK * RVK);

  // helpers: pick NT=4 (64-wide wave blocks) when N allows, else NT=2
  auto gemmMix = [&](const float* coef, const unsigned short* W, float* Dst,
                     const float* bias, int Ncols, int EPI) {
    if (Ncols % 64 == 0) {
      int groups = (NNK / 16) * (Ncols / 64);
      gemm_bf16_wmma<true, 4><<<(groups + 3) / 4, 128, 0, stream>>>(
          x, coef, W, Dst, bias, nullptr, nullptr, NNK, CCK, Ncols, EPI);
    } else {
      int groups = (NNK / 16) * (Ncols / 32);
      gemm_bf16_wmma<true, 2><<<(groups + 3) / 4, 128, 0, stream>>>(
          x, coef, W, Dst, bias, nullptr, nullptr, NNK, CCK, Ncols, EPI);
    }
  };
  auto gemmPlain = [&](const float* A, const unsigned short* W, float* Dst,
                       const float* bias, const float* aux0, const float* aux1,
                       int K, int Ncols, int EPI) {
    if (Ncols % 64 == 0) {
      int groups = (NNK / 16) * (Ncols / 64);
      gemm_bf16_wmma<false, 4><<<(groups + 3) / 4, 128, 0, stream>>>(
          A, nullptr, W, Dst, bias, aux0, aux1, NNK, K, Ncols, EPI);
    } else {
      int groups = (NNK / 16) * (Ncols / 32);
      gemm_bf16_wmma<false, 2><<<(groups + 3) / 4, 128, 0, stream>>>(
          A, nullptr, W, Dst, bias, aux0, aux1, NNK, K, Ncols, EPI);
    }
  };

  // projections (mix fused into A-fragment build)
  gemmMix(x_r, WrB, rbuf, nullptr, CCK, EPI_NONE);                 // r
  gemmMix(x_k, WkB, kbuf, nullptr, CCK, EPI_NONE);                 // k (raw)
  gemmMix(x_v, WvB, vbuf, nullptr, CCK, EPI_NONE);                 // v (raw)

  // v-gate LoRA, lerp with v_first fused in epilogue
  gemmMix(x_v, v2B, vgh, nullptr, RVK, EPI_NONE);
  gemmPlain(vgh, v1B, vbuf, v0, vbuf, vfirst, RVK, CCK, EPI_VLERP);

  // w LoRA -> decay, fully fused (tanh inner, softplus/exp outer)
  gemmMix(x_w, w2B, wh, nullptr, RWK, EPI_TANH);
  gemmPlain(wh, w1B, decbuf, w0, nullptr, nullptr, RWK, CCK, EPI_WDEC);

  // a LoRA -> a_gate
  gemmMix(x_a, a2B, ah, nullptr, RAK, EPI_NONE);
  gemmPlain(ah, a1B, agbuf, a0, nullptr, nullptr, RAK, CCK, EPI_SIGB);

  // g LoRA (sigmoid inner act)
  gemmMix(x_g, g2B, gh, nullptr, RGK, EPI_SIG);
  gemmPlain(gh, g1B, gbuf, nullptr, nullptr, nullptr, RGK, CCK, EPI_NONE);

  // kk normalization + k modification
  prep_scan<<<NNK * HHK, DDK, 0, stream>>>(kbuf, agbuf, k_k, k_a, kkbuf);

  // sequential WKV scan (writes y4 and final state sT)
  wkv_scan<<<BB * HHK, DDK, 0, stream>>>(rbuf, kbuf, vbuf, decbuf, kkbuf, agbuf,
                                         s0, y4buf, outST);

  // groupnorm + correction + gate (decay buffer reused as output)
  gn_corr_gate<<<NNK * HHK, DDK, 0, stream>>>(y4buf, rbuf, kbuf, vbuf, gbuf,
                                              r_k, gnw, gnb, decbuf);

  // final projection into d_out
  gemmPlain(decbuf, WoB, outY, nullptr, nullptr, nullptr, CCK, CCK, EPI_NONE);

  // v_first pass-through
  hipMemcpyAsync(outVF, vfirst, NCK * sizeof(float), hipMemcpyDeviceToDevice, stream);
}
